// EncoderLayer_73143293051508
// MI455X (gfx1250) — matrix operand order repaired and checked by reading, not compiled, not run on hardware
//
#include <hip/hip_runtime.h>

// ---------------------------------------------------------------------------
// Transformer encoder layer for MI455X (gfx1250, wave32, WMMA bf16)
// B=2, S=2048, D=1024, H=16, hd=64, FFN=4096
// ---------------------------------------------------------------------------

#define D_MODEL 1024
#define SEQ     2048
#define BATCH   2
#define NHEAD   16
#define HDIM    64
#define FFN_H   4096
#define ROWS    (BATCH * SEQ)      // 4096 token rows
#define LN_EPS  1e-5f

typedef __attribute__((ext_vector_type(16))) __bf16 v16bf;
typedef __attribute__((ext_vector_type(8)))  float  v8f;

union FragBF16 {
    v16bf v;
    uint4 q[2];
};

__device__ __forceinline__ v8f wmma_bf16f32(v16bf a, v16bf b, v8f c) {
    // D = A(16x32 bf16) * B(32x16 bf16) + C(16x16 f32)
    return __builtin_amdgcn_wmma_f32_16x16x32_bf16(
        /*neg_a=*/false, a, /*neg_b=*/false, b,
        /*c_mod=*/(short)0, c, /*reuse_a=*/false, /*reuse_b=*/false);
}

// --- gfx1250 async global->LDS copy (ASYNCcnt-tracked), with fallback -------
#if defined(__HIP_DEVICE_COMPILE__) && __has_builtin(__builtin_amdgcn_global_load_async_to_lds_b128)
#define HAS_ASYNC_LDS 1
#else
#define HAS_ASYNC_LDS 0
#endif

#if HAS_ASYNC_LDS
// builtin signature: (v4i addrspace(1)* src, v4i addrspace(3)* dst, Imm off, Imm cpol)
typedef int v4i32 __attribute__((vector_size(16)));
typedef __attribute__((address_space(1))) v4i32 as1_v4i32;
typedef __attribute__((address_space(3))) v4i32 as3_v4i32;
#define GPTR(p) ((as1_v4i32*)(p))
#define LPTR(p) ((as3_v4i32*)(p))
template <int N>
__device__ __forceinline__ void wait_asynccnt() {
#if __has_builtin(__builtin_amdgcn_s_wait_asynccnt)
    __builtin_amdgcn_s_wait_asynccnt(N);
#else
    asm volatile("s_wait_asynccnt %0" ::"n"(N));
#endif
}
#endif

// ---------------------------------------------------------------------------
// Elementwise f32 -> bf16 convert
// ---------------------------------------------------------------------------
__global__ __launch_bounds__(256)
void cvt_f32_to_bf16(const float* __restrict__ in, __bf16* __restrict__ out, int n) {
    int i = blockIdx.x * 256 + threadIdx.x;
    if (i < n) out[i] = (__bf16)in[i];
}

// ---------------------------------------------------------------------------
// Transpose [K][N] f32 -> [N][K] bf16  (weights become N-major, K contiguous)
// ---------------------------------------------------------------------------
__global__ __launch_bounds__(256)
void transpose_to_bf16(const float* __restrict__ W, __bf16* __restrict__ Wt,
                       int K, int N) {
    int i = blockIdx.x * 256 + threadIdx.x;
    if (i < N * K) {
        int n = i / K;
        int k = i - n * K;
        Wt[i] = (__bf16)W[(size_t)k * N + n];
    }
}

// ---------------------------------------------------------------------------
// Generic WMMA GEMM:  C[M][N] = A[M][K](bf16, row-major) * Bt[N][K](bf16)
//                      + bias[N]  (+ resid[M][N])  (ReLU)  -> f32 or bf16
// Block tile 128x128, K-step 32. 256 threads = 8 waves; each wave does a
// 32x64 sub-tile = 2x4 WMMA accumulators.
// Tiles are staged with double-buffered GLOBAL_LOAD_ASYNC_TO_LDS_B128
// (ASYNCcnt) so copy of tile k+1 overlaps WMMA of tile k.
// ---------------------------------------------------------------------------
template <bool RELU, bool RESID, bool OUT_BF16>
__global__ __launch_bounds__(256)
void gemm_bf16_kernel(const __bf16* __restrict__ A, const __bf16* __restrict__ Bt,
                      const float* __restrict__ bias, const float* __restrict__ resid,
                      float* __restrict__ Cf, __bf16* __restrict__ Cb,
                      int M, int N, int K) {
    __shared__ __bf16 sA[2][128 * 32];
    __shared__ __bf16 sB[2][128 * 32];

    const int t    = threadIdx.x;
    const int lane = t & 31;
    const int wave = t >> 5;
    const int half = lane >> 4;        // 0: lanes 0-15, 1: lanes 16-31
    const int l16  = lane & 15;

    const int wm = (wave & 3) * 32;    // wave row offset within block tile
    const int wn = (wave >> 2) * 64;   // wave col offset within block tile
    const int bm = blockIdx.x * 128;
    const int bn = blockIdx.y * 128;

    // tile cooperative load mapping: thread -> (row, 16-elem k chunk)
    const int lr = t >> 1;             // 0..127
    const int lk = (t & 1) * 16;       // 0 or 16

    const __bf16* gaBase = A  + (size_t)(bm + lr) * K + lk;
    const __bf16* gbBase = Bt + (size_t)(bn + lr) * K + lk;
    const int ldsOff = lr * 32 + lk;

    v8f acc[2][4];
    for (int mi = 0; mi < 2; ++mi)
        for (int ni = 0; ni < 4; ++ni)
            for (int e = 0; e < 8; ++e) acc[mi][ni][e] = 0.0f;

#if HAS_ASYNC_LDS
    // ---- async double-buffered pipeline -----------------------------------
    auto issue_tile = [&](int k0, int buf) {
        const __bf16* ga = gaBase + k0;
        const __bf16* gb = gbBase + k0;
        __bf16* la = &sA[buf][ldsOff];
        __bf16* lb = &sB[buf][ldsOff];
        __builtin_amdgcn_global_load_async_to_lds_b128(GPTR(ga), LPTR(la), 0,  0);
        __builtin_amdgcn_global_load_async_to_lds_b128(GPTR(ga), LPTR(la), 16, 0);
        __builtin_amdgcn_global_load_async_to_lds_b128(GPTR(gb), LPTR(lb), 0,  0);
        __builtin_amdgcn_global_load_async_to_lds_b128(GPTR(gb), LPTR(lb), 16, 0);
    };

    issue_tile(0, 0);
    for (int k0 = 0; k0 < K; k0 += 32) {
        const int cur = (k0 >> 5) & 1;
        if (k0 + 32 < K) {
            issue_tile(k0 + 32, cur ^ 1);      // overlap copy of next tile
            wait_asynccnt<4>();                // current tile's 4 copies landed
        } else {
            wait_asynccnt<0>();
        }
        __syncthreads();                        // all waves' copies visible

        FragBF16 af[2];
        const int kb8 = half * 8;
        for (int mi = 0; mi < 2; ++mi) {
            const int row = wm + mi * 16 + l16;
            af[mi].q[0] = *(const uint4*)&sA[cur][row * 32 + kb8];
            af[mi].q[1] = *(const uint4*)&sA[cur][row * 32 + 16 + kb8];
        }
        FragBF16 bfrag[4];
        const int kb16 = half * 16;
        for (int ni = 0; ni < 4; ++ni) {
            const int col = wn + ni * 16 + l16;
            bfrag[ni].q[0] = *(const uint4*)&sB[cur][col * 32 + kb8];
            bfrag[ni].q[1] = *(const uint4*)&sB[cur][col * 32 + 16 + kb8];
        }
        for (int mi = 0; mi < 2; ++mi)
            for (int ni = 0; ni < 4; ++ni)
                acc[mi][ni] = wmma_bf16f32(af[mi].v, bfrag[ni].v, acc[mi][ni]);
        // buffer `cur` is refilled (tile k+2) by the next iteration's issue_tile, before
        // that iteration's barrier, so no barrier orders the refill after the other
        // waves' ds reads of this tile.
    }
#else
    // ---- fallback: synchronous VGPR staging -------------------------------
    for (int k0 = 0; k0 < K; k0 += 32) {
        const __bf16* ga = gaBase + k0;
        const __bf16* gb = gbBase + k0;
        uint4 a0 = *(const uint4*)(ga);
        uint4 a1 = *(const uint4*)(ga + 8);
        uint4 b0 = *(const uint4*)(gb);
        uint4 b1 = *(const uint4*)(gb + 8);
        if (k0 + 32 < K) {
            __builtin_prefetch(ga + 32, 0, 0);
            __builtin_prefetch(gb + 32, 0, 0);
        }
        __syncthreads();
        *(uint4*)&sA[0][ldsOff]     = a0;
        *(uint4*)&sA[0][ldsOff + 8] = a1;
        *(uint4*)&sB[0][ldsOff]     = b0;
        *(uint4*)&sB[0][ldsOff + 8] = b1;
        __syncthreads();

        FragBF16 af[2];
        const int kb8 = half * 8;
        for (int mi = 0; mi < 2; ++mi) {
            const int row = wm + mi * 16 + l16;
            af[mi].q[0] = *(const uint4*)&sA[0][row * 32 + kb8];
            af[mi].q[1] = *(const uint4*)&sA[0][row * 32 + 16 + kb8];
        }
        FragBF16 bfrag[4];
        const int kb16 = half * 16;
        for (int ni = 0; ni < 4; ++ni) {
            const int col = wn + ni * 16 + l16;
            bfrag[ni].q[0] = *(const uint4*)&sB[0][col * 32 + kb8];
            bfrag[ni].q[1] = *(const uint4*)&sB[0][col * 32 + 16 + kb8];
        }
        for (int mi = 0; mi < 2; ++mi)
            for (int ni = 0; ni < 4; ++ni)
                acc[mi][ni] = wmma_bf16f32(af[mi].v, bfrag[ni].v, acc[mi][ni]);
    }
#endif

    // epilogue: C layout -> lane holds col l16, rows v + 8*half (v = 0..7)
    for (int mi = 0; mi < 2; ++mi) {
        for (int ni = 0; ni < 4; ++ni) {
            const int col = bn + wn + ni * 16 + l16;
            const float bv = bias[col];
            for (int v = 0; v < 8; ++v) {
                const int row = bm + wm + mi * 16 + v + 8 * half;
                float val = acc[mi][ni][v] + bv;
                if (RESID) val += resid[(size_t)row * N + col];
                if (RELU)  val = fmaxf(val, 0.0f);
                if (OUT_BF16) Cb[(size_t)row * N + col] = (__bf16)val;
                else          Cf[(size_t)row * N + col] = val;
            }
        }
    }
}

// ---------------------------------------------------------------------------
// Repack fused QKV bf16 [ROWS][3*D] into per-head Q[s][d], K[s][d], V^T[d][s]
// column c -> head = c/192, r = c%192, type = r/64, d = r%64
// ---------------------------------------------------------------------------
__global__ __launch_bounds__(256)
void repack_qkv(const __bf16* __restrict__ qkv,
                __bf16* __restrict__ Q, __bf16* __restrict__ Kk,
                __bf16* __restrict__ Vt) {
    size_t i = (size_t)blockIdx.x * 256 + threadIdx.x;
    const size_t total = (size_t)ROWS * 3 * D_MODEL;
    if (i >= total) return;
    int c  = (int)(i % (3 * D_MODEL));
    int bs = (int)(i / (3 * D_MODEL));
    int b  = bs / SEQ;
    int s  = bs - b * SEQ;
    int h  = c / 192;
    int r  = c - h * 192;
    int ty = r >> 6;
    int d  = r & 63;
    const __bf16 val = qkv[i];
    const size_t bh = (size_t)(b * NHEAD + h);
    if (ty == 0)      Q [(bh * SEQ + s) * HDIM + d] = val;
    else if (ty == 1) Kk[(bh * SEQ + s) * HDIM + d] = val;
    else              Vt[(bh * HDIM + d) * SEQ + s] = val;
}

// ---------------------------------------------------------------------------
// Flash-style attention. Grid: (B*H, S/64). 128 threads = 4 waves; each wave
// owns 16 query rows. Online softmax over 32-key blocks; QK^T and PV via WMMA.
// ---------------------------------------------------------------------------
__global__ __launch_bounds__(128)
void attn_kernel(const __bf16* __restrict__ Q, const __bf16* __restrict__ Kk,
                 const __bf16* __restrict__ Vt, __bf16* __restrict__ ctx) {
    __shared__ __bf16 sP[4][16 * 32];      // per-wave P tile (C-layout -> A-layout)

    const int bh   = blockIdx.x;           // 0..31  (b*16 + h)
    const int qt   = blockIdx.y;           // 0..31  (64-row query tile)
    const int t    = threadIdx.x;
    const int lane = t & 31;
    const int wave = t >> 5;
    const int half = lane >> 4;
    const int l16  = lane & 15;
    const int qbase = qt * 64 + wave * 16;

    const __bf16* Qb = Q  + (size_t)bh * SEQ * HDIM;
    const __bf16* Kb = Kk + (size_t)bh * SEQ * HDIM;
    const __bf16* Vb = Vt + (size_t)bh * HDIM * SEQ;
    __bf16* myP = sP[wave];

    // Q fragments (fixed per wave): K-dim = hd = 64 -> 2 chunks of 32
    FragBF16 qf[2];
    {
        const __bf16* qp = Qb + (size_t)(qbase + l16) * HDIM;
        const int kb8 = half * 8;
        for (int dk = 0; dk < 2; ++dk) {
            qf[dk].q[0] = *(const uint4*)(qp + dk * 32 + kb8);
            qf[dk].q[1] = *(const uint4*)(qp + dk * 32 + 16 + kb8);
        }
    }

    v8f o[4];
    for (int nd = 0; nd < 4; ++nd)
        for (int e = 0; e < 8; ++e) o[nd][e] = 0.0f;
    float m[8], l[8];
    for (int r = 0; r < 8; ++r) { m[r] = -1e30f; l[r] = 0.0f; }

    const float scale = 0.125f;            // 1/sqrt(64)
    const int kb16 = half * 16;
    const int kb8  = half * 8;

    for (int kb = 0; kb < SEQ; kb += 32) {
        // ---- scores: S = Q K^T (two 16-key subtiles, K-dim 64) ----
        v8f sacc[2];
        for (int nj = 0; nj < 2; ++nj)
            for (int e = 0; e < 8; ++e) sacc[nj][e] = 0.0f;
        for (int nj = 0; nj < 2; ++nj) {
            const __bf16* kp = Kb + (size_t)(kb + nj * 16 + l16) * HDIM + kb8;
            for (int dk = 0; dk < 2; ++dk) {
                FragBF16 kf;
                kf.q[0] = *(const uint4*)(kp + dk * 32);
                kf.q[1] = *(const uint4*)(kp + dk * 32 + 16);
                sacc[nj] = wmma_bf16f32(qf[dk].v, kf.v, sacc[nj]);
            }
        }
        // ---- online softmax (each lane: 8 rows, 1 col; reduce over 16 lanes)
        float corr[8];
        for (int r = 0; r < 8; ++r) {
            float c0 = sacc[0][r] * scale;
            float c1 = sacc[1][r] * scale;
            float mx = fmaxf(c0, c1);
            mx = fmaxf(mx, __shfl_xor(mx, 1, 32));
            mx = fmaxf(mx, __shfl_xor(mx, 2, 32));
            mx = fmaxf(mx, __shfl_xor(mx, 4, 32));
            mx = fmaxf(mx, __shfl_xor(mx, 8, 32));
            const float mnew = fmaxf(m[r], mx);
            corr[r] = __expf(m[r] - mnew);
            m[r] = mnew;
            float p0 = __expf(c0 - mnew);
            float p1 = __expf(c1 - mnew);
            sacc[0][r] = p0; sacc[1][r] = p1;
            float ps = p0 + p1;
            ps += __shfl_xor(ps, 1, 32);
            ps += __shfl_xor(ps, 2, 32);
            ps += __shfl_xor(ps, 4, 32);
            ps += __shfl_xor(ps, 8, 32);
            l[r] = l[r] * corr[r] + ps;
        }
        for (int nd = 0; nd < 4; ++nd)
            for (int r = 0; r < 8; ++r) o[nd][r] *= corr[r];

        // ---- P (C-layout) -> LDS -> A-layout fragment ----
        for (int nj = 0; nj < 2; ++nj)
            for (int r = 0; r < 8; ++r)
                myP[(r + 8 * half) * 32 + nj * 16 + l16] = (__bf16)sacc[nj][r];
        __syncthreads();
        FragBF16 pf;
        pf.q[0] = *(const uint4*)&myP[l16 * 32 + kb8];
        pf.q[1] = *(const uint4*)&myP[l16 * 32 + 16 + kb8];

        // ---- O += P V   (V^T rows are K-contiguous) ----
        for (int nd = 0; nd < 4; ++nd) {
            FragBF16 vf;
            const __bf16* vp = Vb + (size_t)(nd * 16 + l16) * SEQ + kb + kb8;
            vf.q[0] = *(const uint4*)(vp);
            vf.q[1] = *(const uint4*)(vp + 16);
            o[nd] = wmma_bf16f32(pf.v, vf.v, o[nd]);
        }
        __syncthreads();
    }

    // ---- finalize: O /= l, scatter to context [b*S+s][h*64+d] bf16 ----
    const int h = bh & 15;
    const int b = bh >> 4;
    for (int nd = 0; nd < 4; ++nd) {
        const int col = h * HDIM + nd * 16 + l16;
        for (int r = 0; r < 8; ++r) {
            const int row = qbase + r + 8 * half;
            const float val = o[nd][r] / l[r];
            ctx[((size_t)(b * SEQ + row)) * D_MODEL + col] = (__bf16)val;
        }
    }
}

// ---------------------------------------------------------------------------
// LayerNorm over rows of 1024; optionally also emits bf16 copy.
// ---------------------------------------------------------------------------
__global__ __launch_bounds__(256)
void layernorm_kernel(const float* __restrict__ in, const float* __restrict__ gamma,
                      const float* __restrict__ beta, float* __restrict__ outf,
                      __bf16* __restrict__ outb) {
    const int N = D_MODEL;
    __shared__ float red[8];
    const int row = blockIdx.x;
    const int t   = threadIdx.x;
    const float* r = in + (size_t)row * N;

    float s = 0.0f;
    for (int i = t; i < N; i += 256) s += r[i];
    for (int msk = 16; msk >= 1; msk >>= 1) s += __shfl_xor(s, msk, 32);
    if ((t & 31) == 0) red[t >> 5] = s;
    __syncthreads();
    float tot = 0.0f;
    for (int i = 0; i < 8; ++i) tot += red[i];
    const float mean = tot * (1.0f / N);
    __syncthreads();

    float vs = 0.0f;
    for (int i = t; i < N; i += 256) { float d = r[i] - mean; vs += d * d; }
    for (int msk = 16; msk >= 1; msk >>= 1) vs += __shfl_xor(vs, msk, 32);
    if ((t & 31) == 0) red[t >> 5] = vs;
    __syncthreads();
    float var = 0.0f;
    for (int i = 0; i < 8; ++i) var += red[i];
    const float rsq = rsqrtf(var * (1.0f / N) + LN_EPS);

    for (int i = t; i < N; i += 256) {
        const float val = gamma[i] * (r[i] - mean) * rsq + beta[i];
        outf[(size_t)row * N + i] = val;
        if (outb) outb[(size_t)row * N + i] = (__bf16)val;
    }
}

// ---------------------------------------------------------------------------
// Host orchestration
// ---------------------------------------------------------------------------
extern "C" void kernel_launch(void* const* d_in, const int* in_sizes, int n_in,
                              void* d_out, int out_size, void* d_ws, size_t ws_size,
                              hipStream_t stream) {
    (void)in_sizes; (void)n_in; (void)out_size; (void)ws_size;
    const float* x      = (const float*)d_in[0];
    const float* W_qkv  = (const float*)d_in[1];
    const float* b_qkv  = (const float*)d_in[2];
    const float* W_o    = (const float*)d_in[3];
    const float* b_o    = (const float*)d_in[4];
    const float* gamma1 = (const float*)d_in[5];
    const float* beta1  = (const float*)d_in[6];
    const float* W1     = (const float*)d_in[7];
    const float* b1     = (const float*)d_in[8];
    const float* W2     = (const float*)d_in[9];
    const float* b2     = (const float*)d_in[10];
    const float* gamma2 = (const float*)d_in[11];
    const float* beta2  = (const float*)d_in[12];

    char* ws = (char*)d_ws;
    size_t off = 0;
    const size_t XBF  = off; off += (size_t)ROWS * D_MODEL * 2;        // x bf16         8M
    const size_t WQKV = off; off += (size_t)3 * D_MODEL * D_MODEL * 2; // Wqkv^T bf16    6M
    const size_t WOT  = off; off += (size_t)D_MODEL * D_MODEL * 2;     // Wo^T bf16      2M
    const size_t W1T  = off; off += (size_t)FFN_H * D_MODEL * 2;       // W1^T bf16      8M
    const size_t W2T  = off; off += (size_t)D_MODEL * FFN_H * 2;       // W2^T bf16      8M
    const size_t QKVB = off; off += (size_t)ROWS * 3 * D_MODEL * 2;    // qkv bf16      24M
    const size_t QH   = off; off += (size_t)ROWS * D_MODEL * 2;        // Q per-head     8M
    const size_t KH   = off; off += (size_t)ROWS * D_MODEL * 2;        // K per-head     8M
    const size_t VTH  = off; off += (size_t)ROWS * D_MODEL * 2;        // V^T per-head   8M
    const size_t CTX  = off; off += (size_t)ROWS * D_MODEL * 2;        // context bf16   8M
    const size_t HF   = off; off += (size_t)ROWS * D_MODEL * 4;        // h f32         16M
    const size_t HB   = off; off += (size_t)ROWS * D_MODEL * 2;        // h bf16         8M
    // aliased regions (lifetimes disjoint):
    const size_t ATTN_RES = QKVB;  // 16M f32 fits in the 24M qkv buffer (qkv dead)
    const size_t FFN1     = QH;    // 32M bf16 spans QH..CTX (all dead after attention)
    const size_t FFN_RES  = XBF;   // 16M f32 spans XBF..WOT (dead after GEMM2)

    __bf16* xbf    = (__bf16*)(ws + XBF);
    __bf16* wqkv_t = (__bf16*)(ws + WQKV);
    __bf16* wo_t   = (__bf16*)(ws + WOT);
    __bf16* w1_t   = (__bf16*)(ws + W1T);
    __bf16* w2_t   = (__bf16*)(ws + W2T);
    __bf16* qkvb   = (__bf16*)(ws + QKVB);
    __bf16* qh     = (__bf16*)(ws + QH);
    __bf16* kh     = (__bf16*)(ws + KH);
    __bf16* vth    = (__bf16*)(ws + VTH);
    __bf16* ctx    = (__bf16*)(ws + CTX);
    float*  hf     = (float*)(ws + HF);
    __bf16* hb     = (__bf16*)(ws + HB);
    float*  attn_res = (float*)(ws + ATTN_RES);
    __bf16* ffn1     = (__bf16*)(ws + FFN1);
    float*  ffn_res  = (float*)(ws + FFN_RES);

    // 1) convert activations / weights to bf16 (weights transposed to N-major)
    {
        int n = ROWS * D_MODEL;
        cvt_f32_to_bf16<<<(n + 255) / 256, 256, 0, stream>>>(x, xbf, n);
    }
    transpose_to_bf16<<<(3 * D_MODEL * D_MODEL + 255) / 256, 256, 0, stream>>>(W_qkv, wqkv_t, D_MODEL, 3 * D_MODEL);
    transpose_to_bf16<<<(D_MODEL * D_MODEL + 255) / 256, 256, 0, stream>>>(W_o, wo_t, D_MODEL, D_MODEL);
    transpose_to_bf16<<<(FFN_H * D_MODEL + 255) / 256, 256, 0, stream>>>(W1, w1_t, D_MODEL, FFN_H);
    transpose_to_bf16<<<(D_MODEL * FFN_H + 255) / 256, 256, 0, stream>>>(W2, w2_t, FFN_H, D_MODEL);

    // 2) QKV = x @ W_qkv + b_qkv      [4096 x 3072], bf16 out
    gemm_bf16_kernel<false, false, true><<<dim3(ROWS / 128, (3 * D_MODEL) / 128), 256, 0, stream>>>(
        xbf, wqkv_t, b_qkv, nullptr, nullptr, qkvb, ROWS, 3 * D_MODEL, D_MODEL);

    // 3) repack into per-head Q, K, V^T
    {
        size_t n = (size_t)ROWS * 3 * D_MODEL;
        repack_qkv<<<(unsigned)((n + 255) / 256), 256, 0, stream>>>(qkvb, qh, kh, vth);
    }

    // 4) flash attention -> context bf16 [4096 x 1024]
    attn_kernel<<<dim3(BATCH * NHEAD, SEQ / 64), 128, 0, stream>>>(qh, kh, vth, ctx);

    // 5) attn_out = ctx @ W_o + b_o + x   (f32, pre-LN residual)
    gemm_bf16_kernel<false, true, false><<<dim3(ROWS / 128, D_MODEL / 128), 256, 0, stream>>>(
        ctx, wo_t, b_o, x, attn_res, nullptr, ROWS, D_MODEL, D_MODEL);

    // 6) h = LN1(attn_res)  -> f32 + bf16
    layernorm_kernel<<<ROWS, 256, 0, stream>>>(attn_res, gamma1, beta1, hf, hb);

    // 7) ffn1 = relu(h @ W1 + b1)   [4096 x 4096] bf16
    gemm_bf16_kernel<true, false, true><<<dim3(ROWS / 128, FFN_H / 128), 256, 0, stream>>>(
        hb, w1_t, b1, nullptr, nullptr, ffn1, ROWS, FFN_H, D_MODEL);

    // 8) ffn_res = ffn1 @ W2 + b2 + h   (f32)
    gemm_bf16_kernel<false, true, false><<<dim3(ROWS / 128, D_MODEL / 128), 256, 0, stream>>>(
        ffn1, w2_t, b2, hf, ffn_res, nullptr, ROWS, D_MODEL, FFN_H);

    // 9) out = LN2(ffn_res)
    layernorm_kernel<<<ROWS, 256, 0, stream>>>(ffn_res, gamma2, beta2, (float*)d_out, nullptr);
}
